// PsiQRHAttention_79207786872874
// MI455X (gfx1250) — matrix operand order repaired and checked by reading, not compiled, not run on hardware
//
#include <hip/hip_runtime.h>
#include <hip/hip_bf16.h>
#include <math.h>

// ---------------------------------------------------------------------------
// PsiQRH phasor attention for MI455X (gfx1250, wave32, WMMA f16 16x16x32).
// All heavy math runs through v_wmma_f32_16x16x32_f16 with fp32 accumulate.
// The 536MB scores tensor is never materialized: scores, sum-of-squares and
// the P@Hv product are fused per 64-row q-block (flash-style), since the L2
// row normalization is a post-hoc per-row scalar.
// Tile staging uses GLOBAL_LOAD_ASYNC_TO_LDS_B128 (ASYNCcnt), double-buffered
// in the GEMM so the async unit fills tile k+1 while waves run WMMA on tile k.
// Rc is pre-transposed to d-major (tiled transpose kernel) so every WMMA
// B operand in the attention kernel is K x N row-major in LDS, read by load_b.
// Workspace requirement: ~98 MB (see offsets in kernel_launch).
// ---------------------------------------------------------------------------

typedef _Float16 h16;
typedef __attribute__((ext_vector_type(16))) _Float16 v16h;
typedef __attribute__((ext_vector_type(8)))  float    v8f;
typedef int v4i32 __attribute__((vector_size(16)));

#define B_SZ     2
#define S_SZ     2048
#define DMODEL   1024
#define NHEADS   16
#define HDIM     64
#define NROWS    (B_SZ * S_SZ)   // 4096

#if defined(__gfx1250__) && __has_builtin(__builtin_amdgcn_global_load_async_to_lds_b128)
#define ASYNC_LDS 1
#else
#define ASYNC_LDS 0
#endif

// 16-byte global -> LDS copy (async path when available).
__device__ __forceinline__ void cp16_g2l(h16* lds, const h16* glb) {
#if ASYNC_LDS
  __builtin_amdgcn_global_load_async_to_lds_b128((v4i32*)glb, (v4i32*)lds, 0, 0);
#else
  *(uint4*)lds = *(const uint4*)glb;
#endif
}

__device__ __forceinline__ void wait_async_le4() {
#if ASYNC_LDS
#if __has_builtin(__builtin_amdgcn_s_wait_asynccnt)
  __builtin_amdgcn_s_wait_asynccnt((short)4);
#else
  asm volatile("s_wait_asynccnt 0x4" ::: "memory");
#endif
#endif
}
__device__ __forceinline__ void wait_async_0() {
#if ASYNC_LDS
#if __has_builtin(__builtin_amdgcn_s_wait_asynccnt)
  __builtin_amdgcn_s_wait_asynccnt((short)0);
#else
  asm volatile("s_wait_asynccnt 0x0" ::: "memory");
#endif
#endif
}

// ---------------- WMMA helpers (CDNA5 layouts) -----------------------------

__device__ __forceinline__ v8f wmma_f16(v16h a, v16h b, v8f c) {
  // (neg_a, A, neg_b, B, c_mod, C, reuse_a, reuse_b)
  return __builtin_amdgcn_wmma_f32_16x16x32_f16(false, a, false, b,
                                                (short)0, c, false, false);
}

// A fragment: 16x32 f16, M x K. Lane (0..15): M=lane, K 0..7 in v0..3 and
// K 16..23 in v4..7; lanes 16..31 hold K 8..15 / 24..31 for the same rows.
__device__ __forceinline__ v16h load_a(const h16* base, int ld) {
  const int lane = threadIdx.x & 31;
  const int m = lane & 15, hh = lane >> 4;
  const h16* p0 = base + m * ld + hh * 8;
  const h16* p1 = base + m * ld + 16 + hh * 8;
  v16h r;
#pragma unroll
  for (int i = 0; i < 8; ++i) r[i] = p0[i];
#pragma unroll
  for (int i = 0; i < 8; ++i) r[8 + i] = p1[i];
  return r;
}

// B fragment: 32x16 f16, K x N. Lane (0..15): N=lane, K 0..7 in v0..3 and
// K 16..23 in v4..7; lanes 16..31 hold K 8..15 / 24..31 for the same columns.
__device__ __forceinline__ v16h load_b(const h16* base, int ld) {
  const int lane = threadIdx.x & 31;
  const int n = lane & 15, hh = lane >> 4;
  const h16* p0 = base + (hh * 8) * ld + n;
  const h16* p1 = base + (16 + hh * 8) * ld + n;
  v16h r;
#pragma unroll
  for (int i = 0; i < 8; ++i) r[i] = p0[i * ld];
#pragma unroll
  for (int i = 0; i < 8; ++i) r[8 + i] = p1[i * ld];
  return r;
}

// ---------------- fp32 -> f16 conversion -----------------------------------

__global__ void cvt_f32_f16_kernel(const float* __restrict__ x,
                                   h16* __restrict__ y, int n) {
  int i = blockIdx.x * 256 + threadIdx.x;
  if (i < n) y[i] = (h16)x[i];
}

// ---------------- generic tiled WMMA GEMM (double-buffered, async LDS) -----
// C[M,N] = A[M,K] @ W[K,N] + bias  ; A,W row-major f16; out f16 or f32.
// Block tile 128x128, BK=32, 256 threads = 8 waves; wave (wm in 0..3, wn in
// 0..1) owns a 32x64 sub-tile = 2x4 WMMA accumulators.

template <int OUTF32>
__global__ void __launch_bounds__(256)
gemm_f16_kernel(const h16* __restrict__ A, const h16* __restrict__ W,
                const float* __restrict__ bias,
                h16* __restrict__ C16, float* __restrict__ C32,
                int M, int N, int K) {
  __shared__ __align__(16) h16 As[2][128 * 40];   // lda=40 (pad 8)
  __shared__ __align__(16) h16 Bs[2][32 * 136];   // ldb=136 (pad 8)
  const int t = threadIdx.x;
  const int wave = t >> 5, lane = t & 31;
  const int wm = wave >> 1, wn = wave & 1;
  const int m0 = blockIdx.y * 128;
  const int n0 = blockIdx.x * 128;

  v8f acc[2][4];
#pragma unroll
  for (int i = 0; i < 2; ++i)
#pragma unroll
    for (int j = 0; j < 4; ++j)
#pragma unroll
      for (int r = 0; r < 8; ++r) acc[i][j][r] = 0.0f;

  // stage one 128x32 A tile + 32x128 W tile into buffer `buf` (4 ops/thread)
  auto stage = [&](int buf, int k0) {
#pragma unroll
    for (int it = 0; it < 2; ++it) {
      int r = (t >> 2) + it * 64, c8 = (t & 3) * 8;
      cp16_g2l(&As[buf][r * 40 + c8], &A[(size_t)(m0 + r) * K + k0 + c8]);
    }
#pragma unroll
    for (int it = 0; it < 2; ++it) {
      int idx = t + it * 256;
      int kr = idx >> 4, c8 = (idx & 15) * 8;
      cp16_g2l(&Bs[buf][kr * 136 + c8], &W[(size_t)(k0 + kr) * N + n0 + c8]);
    }
  };

  const int nK = K >> 5;
  stage(0, 0);
  for (int kt = 0; kt < nK; ++kt) {
    const int cur = kt & 1;
    if (kt + 1 < nK) {
      stage(cur ^ 1, (kt + 1) << 5);   // async fill of the other buffer
      wait_async_le4();                // current tile's 4 ops complete
    } else {
      wait_async_0();
    }
    __syncthreads();

    v16h af0 = load_a(&As[cur][(wm * 32) * 40], 40);
    v16h af1 = load_a(&As[cur][(wm * 32 + 16) * 40], 40);
#pragma unroll
    for (int j = 0; j < 4; ++j) {
      v16h bf = load_b(&Bs[cur][wn * 64 + j * 16], 136);
      acc[0][j] = wmma_f16(af0, bf, acc[0][j]);
      acc[1][j] = wmma_f16(af1, bf, acc[1][j]);
    }
    __syncthreads();   // all waves done with buf[cur] before it is re-staged
  }

  const int nl = lane & 15, mh = lane >> 4;
  float bj[4];
#pragma unroll
  for (int j = 0; j < 4; ++j) bj[j] = bias[n0 + wn * 64 + j * 16 + nl];
#pragma unroll
  for (int i = 0; i < 2; ++i)
#pragma unroll
    for (int j = 0; j < 4; ++j)
#pragma unroll
      for (int r = 0; r < 8; ++r) {
        int m = m0 + wm * 32 + i * 16 + r + 8 * mh;
        int n = n0 + wn * 64 + j * 16 + nl;
        float v = acc[i][j][r] + bj[j];
        if constexpr (OUTF32 != 0) {
          C32[(size_t)m * N + n] = v;
        } else {
          C16[(size_t)m * N + n] = (h16)v;
        }
      }
}

// ---------------- LayerNorm over D_LATENT=1024 -----------------------------

__global__ void __launch_bounds__(256)
layernorm_kernel(const h16* __restrict__ Zp, const float* __restrict__ g,
                 const float* __restrict__ bb, h16* __restrict__ Z) {
  const int row = blockIdx.x;
  const int t = threadIdx.x;
  const h16* x = Zp + (size_t)row * DMODEL;
  __shared__ float red[256];

  float vals[4];
  float s = 0.f;
#pragma unroll
  for (int i = 0; i < 4; ++i) { vals[i] = (float)x[t + i * 256]; s += vals[i]; }
  red[t] = s; __syncthreads();
  for (int o = 128; o > 0; o >>= 1) {
    if (t < o) red[t] += red[t + o];
    __syncthreads();
  }
  float mu = red[0] * (1.0f / 1024.0f);
  __syncthreads();
  float vs = 0.f;
#pragma unroll
  for (int i = 0; i < 4; ++i) { float d = vals[i] - mu; vs += d * d; }
  red[t] = vs; __syncthreads();
  for (int o = 128; o > 0; o >>= 1) {
    if (t < o) red[t] += red[t + o];
    __syncthreads();
  }
  float inv = rsqrtf(red[0] * (1.0f / 1024.0f) + 1e-5f);
  h16* out = Z + (size_t)row * DMODEL;
#pragma unroll
  for (int i = 0; i < 4; ++i) {
    int c = t + i * 256;
    out[c] = (h16)((vals[i] - mu) * inv * g[c] + bb[c]);
  }
}

// ---------------- phasor: phi = X Wphi + bphi; Xc = [X cos, X sin] ---------
// T: [4096, 1024] f16 (heads interleaved).  Xc: [B*H, S, 128] f16.

__global__ void __launch_bounds__(64)
phasor_kernel(const h16* __restrict__ T, const float* __restrict__ Wphi,
              const float* __restrict__ bphi, h16* __restrict__ Xc) {
  const int row = blockIdx.x;      // 0..4095
  const int h = blockIdx.y;        // 0..15
  const int d = threadIdx.x;       // 0..63
  __shared__ float q[64];
  float v = (float)T[(size_t)row * DMODEL + h * HDIM + d];
  q[d] = v;
  __syncthreads();
  float phi = bphi[d];
#pragma unroll 8
  for (int k = 0; k < 64; ++k) phi += q[k] * Wphi[k * 64 + d];
  const int b = row >> 11, s = row & 2047;
  size_t o = (((size_t)(b * NHEADS + h)) * S_SZ + s) * 128 + d;
  Xc[o]      = (h16)(v * cosf(phi));
  Xc[o + 64] = (h16)(v * sinf(phi));
}

// ---------------- tiled transpose: Rc [bh][s][128] -> Rct [bh][d][2048] ----
// 32x32 tiles via padded LDS; coalesced reads and writes.

__global__ void __launch_bounds__(256)
transpose_rc_kernel(const h16* __restrict__ Rc, h16* __restrict__ Rct) {
  __shared__ h16 tile[32][33];
  const int t = threadIdx.x;
  const int tx = t & 31, ty = t >> 5;          // 32 x 8
  const int s0 = blockIdx.x * 32;              // 0..2047
  const int d0 = blockIdx.y * 32;              // 0..127
  const int bh = blockIdx.z;
  const h16* src = Rc + (size_t)bh * S_SZ * 128;
  h16* dst = Rct + (size_t)bh * 128 * S_SZ;
#pragma unroll
  for (int i = 0; i < 32; i += 8)
    tile[ty + i][tx] = src[(size_t)(s0 + ty + i) * 128 + d0 + tx];
  __syncthreads();
#pragma unroll
  for (int i = 0; i < 32; i += 8)
    dst[(size_t)(d0 + ty + i) * S_SZ + s0 + tx] = tile[tx][ty + i];
}

// ---------------- relayout H: [B,S,H,64] -> [B*H, S, 64] -------------------

__global__ void relayout_hv_kernel(const h16* __restrict__ T,
                                   h16* __restrict__ Hv) {
  int idx = blockIdx.x * 256 + threadIdx.x;          // over 4096*1024
  int d = idx & 63, h = (idx >> 6) & 15, row = idx >> 10;
  int b = row >> 11, s = row & 2047;
  Hv[(((size_t)(b * NHEADS + h)) * S_SZ + s) * 64 + d] = T[idx];
}

// ---------------- fused phasor attention -----------------------------------
// Per (q-block of 64, b*h): P = Qc Rc^T (K=128), ssq += sum_k P^2,
// AO += P @ Hv (via LDS round-trip of P), final AO *= 1/max(sqrt(ssq),eps).
// Rct is d-major so the scores B operand loads contiguously.

__global__ void __launch_bounds__(256)
attn_kernel(const h16* __restrict__ Qc, const h16* __restrict__ Rct,
            const h16* __restrict__ Hv, h16* __restrict__ AO) {
  __shared__ __align__(16) h16 qT[64 * 136];    // [q][d]
  __shared__ __align__(16) h16 rtT[128 * 72];   // [d][kcol]
  __shared__ __align__(16) h16 hT[64 * 72];     // [k][d]
  __shared__ __align__(16) h16 pT[64 * 72];     // [q][k] f16 scores
  __shared__ float ssq[64];

  const int t = threadIdx.x;
  const int wave = t >> 5, lane = t & 31;
  const int wm = wave >> 1, wn = wave & 1;   // wave owns 16 rows x 32 cols
  const int q0 = blockIdx.x * 64;
  const int bh = blockIdx.y;

  const h16* Qb  = Qc  + (size_t)bh * S_SZ * 128;
  const h16* Rtb = Rct + (size_t)bh * 128 * S_SZ;
  const h16* Hb  = Hv  + (size_t)bh * S_SZ * 64;

  // async-load q tile once: 64 x 128 (completion covered by first wait below)
#pragma unroll
  for (int it = 0; it < 4; ++it) {
    int idx = t + it * 256;
    int r = idx >> 4, c8 = (idx & 15) * 8;
    cp16_g2l(&qT[r * 136 + c8], &Qb[(size_t)(q0 + r) * 128 + c8]);
  }
  if (t < 64) ssq[t] = 0.f;

  v8f ao[2];
#pragma unroll
  for (int j = 0; j < 2; ++j)
#pragma unroll
    for (int r = 0; r < 8; ++r) ao[j][r] = 0.f;

  for (int kb = 0; kb < S_SZ; kb += 64) {
    __syncthreads();   // prior readers of rtT/hT/pT done; ssq init visible
    // stage Rct block 128(d) x 64(k) and Hv block 64 x 64 via async-to-LDS
#pragma unroll
    for (int it = 0; it < 4; ++it) {
      int idx = t + it * 256;          // 0..1023
      int r = idx >> 3, c8 = (idx & 7) * 8;
      cp16_g2l(&rtT[r * 72 + c8], &Rtb[(size_t)r * S_SZ + kb + c8]);
    }
#pragma unroll
    for (int it = 0; it < 2; ++it) {
      int idx = t + it * 256;
      int r = idx >> 3, c8 = (idx & 7) * 8;
      cp16_g2l(&hT[r * 72 + c8], &Hb[(size_t)(kb + r) * 64 + c8]);
    }
    wait_async_0();
    __syncthreads();

    // scores: P(16x32 per wave) = Qc(q,d) * Rc(k,d)^T, K=128
    v8f p[2];
#pragma unroll
    for (int j = 0; j < 2; ++j)
#pragma unroll
      for (int r = 0; r < 8; ++r) p[j][r] = 0.f;
#pragma unroll
    for (int ds = 0; ds < 4; ++ds) {
      v16h a = load_a(&qT[(wm * 16) * 136 + ds * 32], 136);
#pragma unroll
      for (int j = 0; j < 2; ++j) {
        v16h bf = load_b(&rtT[(ds * 32) * 72 + wn * 32 + j * 16], 72);
        p[j] = wmma_f16(a, bf, p[j]);
      }
    }

    // accumulate sum of squares and spill P (f16) to LDS for re-fragmenting
#pragma unroll
    for (int j = 0; j < 2; ++j)
#pragma unroll
      for (int r = 0; r < 8; ++r) {
        int m = wm * 16 + r + 8 * (lane >> 4);
        int n = wn * 32 + j * 16 + (lane & 15);
        float v = p[j][r];
        atomicAdd(&ssq[m], v * v);
        pT[m * 72 + n] = (h16)v;
      }
    __syncthreads();

    // AO(16x32 per wave) += P(q,k) @ Hv(k,d), K=64
#pragma unroll
    for (int ks = 0; ks < 2; ++ks) {
      v16h a = load_a(&pT[(wm * 16) * 72 + ks * 32], 72);
#pragma unroll
      for (int j = 0; j < 2; ++j) {
        v16h bf = load_b(&hT[(ks * 32) * 72 + wn * 32 + j * 16], 72);
        ao[j] = wmma_f16(a, bf, ao[j]);
      }
    }
  }
  __syncthreads();

  // normalize rows and store back into [B, S, H*64] layout for the Wo GEMM
  const int b = bh >> 4, h = bh & 15;
#pragma unroll
  for (int j = 0; j < 2; ++j)
#pragma unroll
    for (int r = 0; r < 8; ++r) {
      int m = wm * 16 + r + 8 * (lane >> 4);
      int n = wn * 32 + j * 16 + (lane & 15);
      float sc = 1.0f / fmaxf(sqrtf(ssq[m]), 1e-12f);
      AO[((size_t)b * S_SZ + q0 + m) * DMODEL + h * HDIM + n] =
          (h16)(ao[j][r] * sc);
    }
}

// ---------------------------------------------------------------------------

extern "C" void kernel_launch(void* const* d_in, const int* in_sizes, int n_in,
                              void* d_out, int out_size, void* d_ws,
                              size_t ws_size, hipStream_t stream) {
  (void)in_sizes; (void)n_in; (void)out_size; (void)ws_size;
  const float* query = (const float*)d_in[0];
  const float* Wz    = (const float*)d_in[3];
  const float* bz    = (const float*)d_in[4];
  const float* ln_g  = (const float*)d_in[5];
  const float* ln_b  = (const float*)d_in[6];
  const float* Wq    = (const float*)d_in[7];
  const float* bq    = (const float*)d_in[8];
  const float* Wr    = (const float*)d_in[9];
  const float* br    = (const float*)d_in[10];
  const float* Wh    = (const float*)d_in[11];
  const float* bh    = (const float*)d_in[12];
  const float* Wphi  = (const float*)d_in[13];
  const float* bphi  = (const float*)d_in[14];
  const float* Wo    = (const float*)d_in[15];
  const float* bo    = (const float*)d_in[16];
  float* out = (float*)d_out;

  // workspace carve-up (bytes)
  char* ws = (char*)d_ws;
  const size_t MB = 1024 * 1024;
  h16* Wz16  = (h16*)(ws + 0 * MB);
  h16* Wq16  = (h16*)(ws + 2 * MB);
  h16* Wr16  = (h16*)(ws + 4 * MB);
  h16* Wh16  = (h16*)(ws + 6 * MB);
  h16* Wo16  = (h16*)(ws + 8 * MB);
  h16* X16   = (h16*)(ws + 10 * MB);  // query f16; reused later as AO
  h16* Zp16  = (h16*)(ws + 18 * MB);  // pre-LN latent
  h16* Z16   = (h16*)(ws + 26 * MB);  // post-LN latent
  h16* T16   = (h16*)(ws + 34 * MB);  // Q/R/H scratch (sequential reuse)
  h16* Qc16  = (h16*)(ws + 42 * MB);  // [B*H, S, 128]
  h16* Rc16  = (h16*)(ws + 58 * MB);  // [B*H, S, 128]
  h16* Hv16  = (h16*)(ws + 74 * MB);  // [B*H, S, 64]
  h16* Rct16 = (h16*)(ws + 82 * MB);  // [B*H, 128, S] (d-major Rc)
  h16* AO16  = X16;                   // [B, S, 1024] (X16 dead by then)

  const int NW = DMODEL * DMODEL;     // 1M weight elements
  const int NX = NROWS * DMODEL;      // 4M activation elements
  dim3 blk(256);

  // 1) fp32 -> f16 staging
  cvt_f32_f16_kernel<<<(NX + 255) / 256, blk, 0, stream>>>(query, X16, NX);
  cvt_f32_f16_kernel<<<(NW + 255) / 256, blk, 0, stream>>>(Wz, Wz16, NW);
  cvt_f32_f16_kernel<<<(NW + 255) / 256, blk, 0, stream>>>(Wq, Wq16, NW);
  cvt_f32_f16_kernel<<<(NW + 255) / 256, blk, 0, stream>>>(Wr, Wr16, NW);
  cvt_f32_f16_kernel<<<(NW + 255) / 256, blk, 0, stream>>>(Wh, Wh16, NW);
  cvt_f32_f16_kernel<<<(NW + 255) / 256, blk, 0, stream>>>(Wo, Wo16, NW);

  dim3 ggrid(DMODEL / 128, NROWS / 128);  // (8, 32)

  // 2) Z-pre = query @ Wz + bz ; 3) LayerNorm
  gemm_f16_kernel<0><<<ggrid, blk, 0, stream>>>(X16, Wz16, bz, Zp16, nullptr,
                                                NROWS, DMODEL, DMODEL);
  layernorm_kernel<<<NROWS, blk, 0, stream>>>(Zp16, ln_g, ln_b, Z16);

  // 4) Q -> Qc ; R -> Rc -> Rct ; H -> Hv
  dim3 pgrid(NROWS, NHEADS);
  gemm_f16_kernel<0><<<ggrid, blk, 0, stream>>>(Z16, Wq16, bq, T16, nullptr,
                                                NROWS, DMODEL, DMODEL);
  phasor_kernel<<<pgrid, dim3(64), 0, stream>>>(T16, Wphi, bphi, Qc16);
  gemm_f16_kernel<0><<<ggrid, blk, 0, stream>>>(Z16, Wr16, br, T16, nullptr,
                                                NROWS, DMODEL, DMODEL);
  phasor_kernel<<<pgrid, dim3(64), 0, stream>>>(T16, Wphi, bphi, Rc16);
  transpose_rc_kernel<<<dim3(S_SZ / 32, 128 / 32, B_SZ * NHEADS), blk, 0,
                        stream>>>(Rc16, Rct16);
  gemm_f16_kernel<0><<<ggrid, blk, 0, stream>>>(Z16, Wh16, bh, T16, nullptr,
                                                NROWS, DMODEL, DMODEL);
  relayout_hv_kernel<<<(NX + 255) / 256, blk, 0, stream>>>(T16, Hv16);

  // 5) fused scores + L2-norm + PV
  dim3 agrid(S_SZ / 64, B_SZ * NHEADS);   // (32, 32)
  attn_kernel<<<agrid, blk, 0, stream>>>(Qc16, Rct16, Hv16, AO16);

  // 6) out = AO @ Wo + bo  (fp32 out)
  gemm_f16_kernel<1><<<ggrid, blk, 0, stream>>>(AO16, Wo16, bo, nullptr, out,
                                                NROWS, DMODEL, DMODEL);
}
